// MultiHeadAttention_75393855914545
// MI455X (gfx1250) — matrix operand order repaired and checked by reading, not compiled, not run on hardware
//
#include <hip/hip_runtime.h>
#include <hip/hip_bf16.h>
#include <stdint.h>

// ---------------------------------------------------------------------------
// MHA forward for MI455X (gfx1250, wave32, WMMA 16x16x32 bf16 -> f32)
// Pipeline: x->bf16 | GEMM Q/K/V (bf16) | V transpose | flash-attn | GEMM out
// CDNA5 paths: v_wmma_f32_16x16x32_bf16, double-buffered
// global_load_async_to_lds_b128 + s_wait_asynccnt, v_permlane16 reductions,
// exp2-domain softmax with WMMA row-sums.
// ---------------------------------------------------------------------------

#define D_MODEL 1024
#define SEQ     2048
#define BATCH   2
#define NHEADS  16
#define DK      64
#define MROWS   (BATCH * SEQ)          // 4096

typedef __attribute__((ext_vector_type(16))) __bf16 v16bf;
typedef __attribute__((ext_vector_type(8)))  float  v8f;

__device__ __forceinline__ __bf16 f2bf(float f) { return (__bf16)f; }  // native cvt

__device__ __forceinline__ unsigned packbf2(float lo, float hi) {
    union { __bf16 b[2]; unsigned u; } p;
    p.b[0] = f2bf(lo); p.b[1] = f2bf(hi);
    return p.u;
}

// ---- async global->LDS copy (ASYNCcnt-tracked, bypasses VGPRs) -------------
__device__ __forceinline__ void async_copy_b128(unsigned lds_addr, const void* gaddr) {
    asm volatile("global_load_async_to_lds_b128 %0, %1, off"
                 :: "v"(lds_addr), "v"((uint64_t)(uintptr_t)gaddr)
                 : "memory");
}
__device__ __forceinline__ void wait_async() {
    asm volatile("s_wait_asynccnt 0x0" ::: "memory");
}

// ---- single-VALU 16-lane xor-butterfly step via v_permlane16_b32 -----------
__host__ __device__ constexpr uint64_t permctl(int k) {
    uint64_t c = 0;
    for (int i = 0; i < 16; ++i) c |= (uint64_t)((i ^ k) & 0xF) << (4 * i);
    return c;
}
template <int K>
__device__ __forceinline__ float permxor16(float x) {
    constexpr uint64_t c = permctl(K);
    int xi = __builtin_bit_cast(int, x);
    int r  = __builtin_amdgcn_permlane16(xi, xi, (int)(c & 0xFFFFFFFFu),
                                         (int)(c >> 32), false, false);
    return __builtin_bit_cast(float, r);
}
__device__ __forceinline__ float groupmax16(float x) {
    x = fmaxf(x, permxor16<1>(x));
    x = fmaxf(x, permxor16<2>(x));
    x = fmaxf(x, permxor16<4>(x));
    x = fmaxf(x, permxor16<8>(x));
    return x;
}

// ---------------------------------------------------------------------------
// fp32 -> bf16 bulk convert (x only), 8 elements / thread
// ---------------------------------------------------------------------------
__global__ __launch_bounds__(256)
void f32_to_bf16_kernel(const float* __restrict__ in, __bf16* __restrict__ out, int n) {
    int i = (blockIdx.x * 256 + threadIdx.x) * 8;
    if (i >= n) return;
    float4 f0 = *(const float4*)(in + i);
    float4 f1 = *(const float4*)(in + i + 4);
    union { unsigned w[4]; uint4 u; } pk;
    pk.w[0] = packbf2(f0.x, f0.y);
    pk.w[1] = packbf2(f0.z, f0.w);
    pk.w[2] = packbf2(f1.x, f1.y);
    pk.w[3] = packbf2(f1.z, f1.w);
    *(uint4*)(out + i) = pk.u;
}

// ---------------------------------------------------------------------------
// C[M,N] = (A_bf16[M,K] @ B_f32[K,N] + bias) * scale
// Block 128x128, BK=64, 8 waves (4M x 2N), 16 WMMA / K-step.
// Double-buffered: A via async DMA; B held in registers one iteration ahead,
// converted and stored as packed k-pairs (b32) into transposed [n][k] LDS.
// ---------------------------------------------------------------------------
template <bool OUT_BF16>
__global__ __launch_bounds__(256)
void gemm_bf16_kernel(const __bf16* __restrict__ A, const float* __restrict__ B,
                      const float* __restrict__ bias, void* __restrict__ Cout,
                      int M, int N, int K, float scale) {
    constexpr int BM = 128, BN = 128, BK = 64, LDA = 72, LDB = 72;
    __shared__ __bf16 As[2][BM * LDA];   // [m][k]
    __shared__ __bf16 Bs[2][BN * LDB];   // [n][k] (transposed)

    const int tid  = threadIdx.x;
    const int lane = tid & 31;
    const int wid  = tid >> 5;
    const int wm   = wid >> 1;
    const int wn   = wid & 1;
    const int lm   = lane & 15;
    const int hi   = lane >> 4;
    const int m0   = blockIdx.y * BM;
    const int n0   = blockIdx.x * BN;

    const int arow = tid >> 1, ac0 = (tid & 1) * 32;   // A stage mapping
    const int kp   = (tid >> 3) << 1;                  // B stage: 2 adjacent k rows
    const int nc0  = (tid & 7) * 16;                   //          x 16 n cols

    v8f acc[2][4] = {};
    float4 breg[8];

    auto asyncA = [&](int kk, int buf) {
        const __bf16* src = A + (size_t)(m0 + arow) * K + kk + ac0;
        unsigned la = (unsigned)(uintptr_t)(&As[buf][arow * LDA + ac0]);
#pragma unroll
        for (int i = 0; i < 4; i++) async_copy_b128(la + 16 * i, src + 8 * i);
    };
    auto loadB = [&](int kk) {
        const float4* s0 = (const float4*)(B + (size_t)(kk + kp)     * N + n0 + nc0);
        const float4* s1 = (const float4*)(B + (size_t)(kk + kp + 1) * N + n0 + nc0);
#pragma unroll
        for (int q = 0; q < 4; q++) { breg[q] = s0[q]; breg[q + 4] = s1[q]; }
    };

    asyncA(0, 0);
    loadB(0);

    const int nit = K / BK;
    for (int it = 0; it < nit; ++it) {
        const int buf = it & 1;
        // convert + store this iteration's B registers as packed k-pairs
#pragma unroll
        for (int q = 0; q < 4; q++) {
            float4 f0 = breg[q], f1 = breg[q + 4];
            int nb = nc0 + 4 * q;
            *(unsigned*)(&Bs[buf][(nb + 0) * LDB + kp]) = packbf2(f0.x, f1.x);
            *(unsigned*)(&Bs[buf][(nb + 1) * LDB + kp]) = packbf2(f0.y, f1.y);
            *(unsigned*)(&Bs[buf][(nb + 2) * LDB + kp]) = packbf2(f0.z, f1.z);
            *(unsigned*)(&Bs[buf][(nb + 3) * LDB + kp]) = packbf2(f0.w, f1.w);
        }
        wait_async();
        __syncthreads();
        if (it + 1 < nit) {               // kick off next tile while computing
            asyncA((it + 1) * BK, buf ^ 1);
            loadB((it + 1) * BK);
        }
        const __bf16* AsB = As[buf];
        const __bf16* BsB = Bs[buf];
#pragma unroll
        for (int t = 0; t < 2; t++) {
            v16bf af[2];
#pragma unroll
            for (int mt = 0; mt < 2; mt++) {
                const __bf16* p = AsB + (wm * 32 + mt * 16 + lm) * LDA + t * 32 + hi * 8;
#pragma unroll
                for (int e = 0; e < 8; e++) { af[mt][e] = p[e]; af[mt][e + 8] = p[e + 16]; }
            }
            v16bf bfr[4];
#pragma unroll
            for (int nt = 0; nt < 4; nt++) {
                const __bf16* p = BsB + (wn * 64 + nt * 16 + lm) * LDB + t * 32 + hi * 8;
#pragma unroll
                for (int e = 0; e < 8; e++) { bfr[nt][e] = p[e]; bfr[nt][e + 8] = p[e + 16]; }
            }
#pragma unroll
            for (int mt = 0; mt < 2; mt++)
#pragma unroll
                for (int nt = 0; nt < 4; nt++)
                    acc[mt][nt] = __builtin_amdgcn_wmma_f32_16x16x32_bf16(
                        false, af[mt], false, bfr[nt], (short)0, acc[mt][nt], false, false);
        }
    }

    // ---- epilogue: one FMA per element with pre-scaled bias ----
#pragma unroll
    for (int nt = 0; nt < 4; nt++) {
        int gc = n0 + wn * 64 + nt * 16 + lm;
        float bvs = bias[gc] * scale;
#pragma unroll
        for (int mt = 0; mt < 2; mt++) {
#pragma unroll
            for (int r = 0; r < 8; r++) {
                int gr = m0 + wm * 32 + mt * 16 + r + 8 * hi;
                float v = fmaf(acc[mt][nt][r], scale, bvs);
                if (OUT_BF16) ((__bf16*)Cout)[(size_t)gr * N + gc] = f2bf(v);
                else          ((float*) Cout)[(size_t)gr * N + gc] = v;
            }
        }
    }
}

// ---------------------------------------------------------------------------
// V [b*s, 1024] -> V^T [b][h][dk][s]  (once; removes per-tile scatter from the
// flash hot loop).  Tile staged with async DMA, written back coalesced.
// ---------------------------------------------------------------------------
__global__ __launch_bounds__(128)
void transpose_v_kernel(const __bf16* __restrict__ Vb, __bf16* __restrict__ Vt) {
    __shared__ __bf16 T[64 * 72];
    const int tid = threadIdx.x;
    const int s0  = blockIdx.x * 64;
    const int bh  = blockIdx.y;
    const int b = bh >> 4, h = bh & 15;
    {
        int row = tid >> 1, c0 = (tid & 1) * 32;
        const __bf16* src = Vb + ((size_t)(b * SEQ) + s0 + row) * D_MODEL + h * DK + c0;
        unsigned la = (unsigned)(uintptr_t)(T + row * 72 + c0);
#pragma unroll
        for (int i = 0; i < 4; i++) async_copy_b128(la + 16 * i, src + 8 * i);
    }
    wait_async();
    __syncthreads();
    int d = tid >> 1, sc = (tid & 1) * 32;
    __bf16* out = Vt + ((size_t)bh * DK + d) * SEQ + s0 + sc;
#pragma unroll
    for (int q = 0; q < 4; q++) {
        union { uint4 u; __bf16 bx[8]; } pk;
#pragma unroll
        for (int j = 0; j < 8; j++) pk.bx[j] = T[(sc + q * 8 + j) * 72 + d];
        ((uint4*)out)[q] = pk.u;
    }
}

// ---------------------------------------------------------------------------
// Flash attention: 128 queries / WG (4 waves x 32 rows), KV tiles of 64,
// double-buffered async staging of K and pre-transposed V.  Scores arrive in
// exp2 domain (Q pre-scaled by log2(e)/8).  Row sums via WMMA with a ones
// matrix.  36 WMMA per KV step per wave.
// ---------------------------------------------------------------------------
__global__ __launch_bounds__(128)
void flash_attn_kernel(const __bf16* __restrict__ Q, const __bf16* __restrict__ Kb,
                       const __bf16* __restrict__ Vt, __bf16* __restrict__ O) {
    constexpr int BKV = 64, LDK = 72, LDV = 72, LDP = 72, NKV = SEQ / BKV;
    __shared__ __bf16 Ks [2][BKV * LDK];    // [kv][dk]
    __shared__ __bf16 Vts[2][DK * LDV];     // [dk][kv]
    __shared__ __bf16 Ps [4 * 16 * LDP];    // per-wave P staging (reused per mt)

    const int tid  = threadIdx.x;
    const int lane = tid & 31;
    const int w    = tid >> 5;
    const int lm   = lane & 15;
    const int hi   = lane >> 4;
    const int bh   = blockIdx.y;
    const int b    = bh >> 4;
    const int h    = bh & 15;
    const int q0   = blockIdx.x * 128 + w * 32;

    // ---- Q fragments (A layout) for both 16-row tiles ----
    v16bf qf[2][2];
#pragma unroll
    for (int mt = 0; mt < 2; mt++) {
        const __bf16* qrow = Q + ((size_t)(b * SEQ) + q0 + mt * 16 + lm) * D_MODEL + h * DK;
#pragma unroll
        for (int t = 0; t < 2; t++) {
            const __bf16* p = qrow + t * 32 + hi * 8;
#pragma unroll
            for (int e = 0; e < 8; e++) { qf[mt][t][e] = p[e]; qf[mt][t][e + 8] = p[e + 16]; }
        }
    }

    v16bf onesf;
#pragma unroll
    for (int e = 0; e < 16; e++) onesf[e] = (__bf16)1.0f;

    float mrow[2][8], lrow[2][8];
#pragma unroll
    for (int mt = 0; mt < 2; mt++)
#pragma unroll
        for (int r = 0; r < 8; r++) { mrow[mt][r] = -1e30f; lrow[mt][r] = 0.0f; }
    v8f o[2][4] = {};

    auto stage = [&](int kb, int buf) {
        int row = tid >> 1, c0 = (tid & 1) * 32;
        const __bf16* ksrc = Kb + ((size_t)(b * SEQ) + kb + row) * D_MODEL + h * DK + c0;
        unsigned la = (unsigned)(uintptr_t)(&Ks[buf][row * LDK + c0]);
#pragma unroll
        for (int i = 0; i < 4; i++) async_copy_b128(la + 16 * i, ksrc + 8 * i);
        const __bf16* vsrc = Vt + ((size_t)bh * DK + row) * SEQ + kb + c0;   // row = d
        unsigned lv = (unsigned)(uintptr_t)(&Vts[buf][row * LDV + c0]);
#pragma unroll
        for (int i = 0; i < 4; i++) async_copy_b128(lv + 16 * i, vsrc + 8 * i);
    };

    stage(0, 0);

    for (int it = 0; it < NKV; ++it) {
        const int buf = it & 1;
        wait_async();
        __syncthreads();
        if (it + 1 < NKV) stage((it + 1) * BKV, buf ^ 1);   // overlap next tile
        const __bf16* KsB  = Ks[buf];
        const __bf16* VtsB = Vts[buf];

#pragma unroll
        for (int mt = 0; mt < 2; mt++) {
            // ---- S = Q K^T : 4 tiles of 16 kv cols ----
            v8f s[4];
#pragma unroll
            for (int nt = 0; nt < 4; nt++) {
                v8f z = {};
#pragma unroll
                for (int t = 0; t < 2; t++) {
                    v16bf kf;
                    const __bf16* p = KsB + (nt * 16 + lm) * LDK + t * 32 + hi * 8;
#pragma unroll
                    for (int e = 0; e < 8; e++) { kf[e] = p[e]; kf[e + 8] = p[e + 16]; }
                    z = __builtin_amdgcn_wmma_f32_16x16x32_bf16(
                            false, qf[mt][t], false, kf, (short)0, z, false, false);
                }
                s[nt] = z;
            }

            // ---- online softmax in exp2 domain ----
            float a8[8];
#pragma unroll
            for (int r = 0; r < 8; r++) {
                float mx = fmaxf(fmaxf(s[0][r], s[1][r]), fmaxf(s[2][r], s[3][r]));
                mx = groupmax16(mx);
                float mnew = fmaxf(mrow[mt][r], mx);
                a8[r] = __builtin_amdgcn_exp2f(mrow[mt][r] - mnew);
                float p0 = __builtin_amdgcn_exp2f(s[0][r] - mnew);
                float p1 = __builtin_amdgcn_exp2f(s[1][r] - mnew);
                float p2 = __builtin_amdgcn_exp2f(s[2][r] - mnew);
                float p3 = __builtin_amdgcn_exp2f(s[3][r] - mnew);
                mrow[mt][r] = mnew;
#pragma unroll
                for (int j = 0; j < 4; j++) o[mt][j][r] *= a8[r];
                __bf16* pr = Ps + (w * 16 + r + 8 * hi) * LDP;
                pr[lm]      = f2bf(p0);
                pr[lm + 16] = f2bf(p1);
                pr[lm + 32] = f2bf(p2);
                pr[lm + 48] = f2bf(p3);
            }

            // ---- P fragments + row sums via WMMA against ones ----
            v16bf pf[2];
#pragma unroll
            for (int t = 0; t < 2; t++) {
                const __bf16* pp = Ps + (w * 16 + lm) * LDP + t * 32 + hi * 8;
#pragma unroll
                for (int e = 0; e < 8; e++) { pf[t][e] = pp[e]; pf[t][e + 8] = pp[e + 16]; }
            }
            v8f zs = {};
            zs = __builtin_amdgcn_wmma_f32_16x16x32_bf16(false, pf[0], false, onesf, (short)0, zs, false, false);
            zs = __builtin_amdgcn_wmma_f32_16x16x32_bf16(false, pf[1], false, onesf, (short)0, zs, false, false);
#pragma unroll
            for (int r = 0; r < 8; r++) lrow[mt][r] = lrow[mt][r] * a8[r] + zs[r];

            // ---- O += P V ----
#pragma unroll
            for (int j = 0; j < 4; j++) {
#pragma unroll
                for (int t = 0; t < 2; t++) {
                    v16bf vf;
                    const __bf16* vp = VtsB + (j * 16 + lm) * LDV + t * 32 + hi * 8;
#pragma unroll
                    for (int e = 0; e < 8; e++) { vf[e] = vp[e]; vf[e + 8] = vp[e + 16]; }
                    o[mt][j] = __builtin_amdgcn_wmma_f32_16x16x32_bf16(
                                   false, pf[t], false, vf, (short)0, o[mt][j], false, false);
                }
            }
        }
    }

    // ---- normalize and write merged-head output (bf16) ----
#pragma unroll
    for (int mt = 0; mt < 2; mt++) {
#pragma unroll
        for (int j = 0; j < 4; j++) {
            int gc = h * DK + j * 16 + lm;
#pragma unroll
            for (int r = 0; r < 8; r++) {
                int gr = q0 + mt * 16 + r + 8 * hi;
                O[((size_t)(b * SEQ) + gr) * D_MODEL + gc] = f2bf(o[mt][j][r] / lrow[mt][r]);
            }
        }
    }
}

// ---------------------------------------------------------------------------
extern "C" void kernel_launch(void* const* d_in, const int* in_sizes, int n_in,
                              void* d_out, int out_size, void* d_ws, size_t ws_size,
                              hipStream_t stream) {
    const float* x  = (const float*)d_in[0];
    const float* wq = (const float*)d_in[1];
    const float* bq = (const float*)d_in[2];
    const float* wk = (const float*)d_in[3];
    const float* bk = (const float*)d_in[4];
    const float* wv = (const float*)d_in[5];
    const float* bv = (const float*)d_in[6];
    const float* wo = (const float*)d_in[7];
    const float* bo = (const float*)d_in[8];
    float* out = (float*)d_out;

    const size_t MN = (size_t)MROWS * D_MODEL;     // 4M elements
    __bf16* Qb = (__bf16*)d_ws;
    __bf16* Kb = Qb + MN;
    __bf16* Vb = Kb + MN;
    __bf16* Ob = Vb + MN;
    __bf16* Xb = Ob + MN;                          // x_bf16; reused as V^T later

    f32_to_bf16_kernel<<<(int)(MN / 2048), 256, 0, stream>>>(x, Xb, (int)MN);

    dim3 gg(D_MODEL / 128, MROWS / 128);           // 8 x 32 workgroups
    // Q scaled by log2(e)/sqrt(dk) so softmax runs natively in exp2 domain
    const float qscale = 0.125f * 1.44269504088896340736f;
    gemm_bf16_kernel<true ><<<gg, 256, 0, stream>>>(Xb, wq, bq, Qb, MROWS, D_MODEL, D_MODEL, qscale);
    gemm_bf16_kernel<true ><<<gg, 256, 0, stream>>>(Xb, wk, bk, Kb, MROWS, D_MODEL, D_MODEL, 1.0f);
    gemm_bf16_kernel<true ><<<gg, 256, 0, stream>>>(Xb, wv, bv, Vb, MROWS, D_MODEL, D_MODEL, 1.0f);

    __bf16* Vt = Xb;                               // x_bf16 no longer needed
    transpose_v_kernel<<<dim3(SEQ / 64, BATCH * NHEADS), 128, 0, stream>>>(Vb, Vt);

    flash_attn_kernel<<<dim3(SEQ / 128, BATCH * NHEADS), 128, 0, stream>>>(Qb, Kb, Vt, Ob);

    gemm_bf16_kernel<false><<<gg, 256, 0, stream>>>(Ob, wo, bo, out, MROWS, D_MODEL, D_MODEL, 1.0f);
}
